// LMUFFTCell_40638980555471
// MI455X (gfx1250) — matrix operand order repaired and checked by reading, not compiled, not run on hardware
//
#include <hip/hip_runtime.h>
#include <hip/hip_bf16.h>

// ---------------------------------------------------------------------------
// LMU-FFT cell for MI455X (gfx1250), wave32 + WMMA bf16.
//
//   u = relu(x @ w_u + b_u)                         (bandwidth-trivial)
//   m = causal conv(u, H)  == im2col GEMM           (6.9e10 bf16 FLOPs)
//   h = relu(x @ w_x + b_x + m)                     (3.4e10 bf16 FLOPs)
//
// conv and x@w_x are FUSED: both accumulate into the same 128x256 f32 tile
// per workgroup, so `m` never touches HBM.  The im2col A operand is read
// through a reversed u window in LDS; its Toeplitz structure lets the
// compiler turn per-tile reads into a sliding register window of b128 LDS
// loads (verified in round-1 asm).
// ---------------------------------------------------------------------------

typedef __attribute__((ext_vector_type(16))) __bf16 v16bf;
typedef __attribute__((ext_vector_type(8)))  float  v8f;

#define SEQ    4096
#define BATCHN 16
#define NIN    1024
#define NH     256     // hidden == memory == 256
#define MT     128     // seq rows per workgroup

static __device__ __forceinline__ __bf16 f2bf(float f) {
  union { float f; unsigned u; } c; c.f = f;
  unsigned r = c.u + 0x7FFFu + ((c.u >> 16) & 1u);   // round-to-nearest-even
  unsigned short h = (unsigned short)(r >> 16);
  return __builtin_bit_cast(__bf16, h);
}
static __device__ __forceinline__ __bf16 bf0() {
  return __builtin_bit_cast(__bf16, (unsigned short)0);
}

union AB16 { unsigned u[8]; v16bf v; };

// ---------------------------------------------------------------------------
// Kernel 1: u[b,s] = relu(x[b,s,:] . w_u + b_u) stored as bf16.
// One wave32 per output row.
// ---------------------------------------------------------------------------
__global__ void lmu_u_kernel(const float* __restrict__ x,
                             const float* __restrict__ w_u,
                             const float* __restrict__ b_u,
                             __bf16* __restrict__ ubf) {
  int row  = blockIdx.x * 8 + (threadIdx.x >> 5);
  int lane = threadIdx.x & 31;
  const float4* xr = (const float4*)(x + (size_t)row * NIN);
  const float4* wr = (const float4*)w_u;
  float s = 0.f;
#pragma unroll
  for (int t = 0; t < 8; ++t) {
    int i = t * 32 + lane;
    float4 a = xr[i], w = wr[i];
    s = fmaf(a.x, w.x, s); s = fmaf(a.y, w.y, s);
    s = fmaf(a.z, w.z, s); s = fmaf(a.w, w.w, s);
  }
#pragma unroll
  for (int off = 16; off > 0; off >>= 1) s += __shfl_down(s, off, 32);
  if (lane == 0) ubf[row] = f2bf(fmaxf(s + b_u[0], 0.f));
}

// ---------------------------------------------------------------------------
// Kernel 2: f32 -> bf16 conversions.  Hb keeps H's [mem][tau] layout (tau
// contiguous == WMMA-B K contiguous).  w_x is transposed to [h][k] so each
// B-lane reads 16 consecutive K values.
// ---------------------------------------------------------------------------
__global__ void lmu_cvt_kernel(const float* __restrict__ H,
                               const float* __restrict__ w_x,
                               __bf16* __restrict__ Hb,
                               __bf16* __restrict__ wxb) {
  int idx = blockIdx.x * 256 + threadIdx.x;        // grid sized to NH*SEQ
  Hb[idx] = f2bf(H[idx]);
  if (idx < NIN * NH) {
    int k = idx >> 8, h = idx & 255;               // w_x is [k][h]
    wxb[h * NIN + k] = f2bf(w_x[idx]);
  }
}

// ---------------------------------------------------------------------------
// Kernel 3: fused conv-GEMM + x@w_x GEMM + bias + relu.
// Workgroup = 256 threads (8 waves).  Tile: 128 seq rows x 256 cols.
// Wave w owns cols [32w, 32w+32): 8 M-subtiles x 2 N-subtiles = 16 WMMA accs.
// ---------------------------------------------------------------------------
__global__ __launch_bounds__(256)
void lmu_fused_kernel(const float* __restrict__ x,
                      const __bf16* __restrict__ ubf,
                      const __bf16* __restrict__ Hb,
                      const __bf16* __restrict__ wxb,
                      const float* __restrict__ b_x,
                      float* __restrict__ out) {
  __shared__ __align__(16) __bf16 sUr[SEQ + 256];   // reversed u window
  __shared__ __align__(16) __bf16 sX[MT * 72];      // 128x64 x-tile, padded

  const int p    = blockIdx.x;          // sequence tile
  const int bb   = blockIdx.y;          // batch
  const int s0   = p * MT;
  const int tid  = threadIdx.x;
  const int wv   = tid >> 5;
  const int lane = tid & 31;
  const int lm   = lane & 15;           // A: M row / B: N col within subtile
  const int hg   = lane >> 4;           // lane group (K offset selector)

  // ---- stage ur[j] = u[b, s0+127-j], zero-padded past the sequence start ---
  const int kmax = s0 + MT;             // conv contraction bound (tau < kmax)
  const __bf16* ub = ubf + (size_t)bb * SEQ;
  for (int j = tid; j < kmax + MT; j += 256) {
    int si = s0 + (MT - 1) - j;
    sUr[j] = (si >= 0) ? ub[si] : bf0();
  }
  __syncthreads();

  v8f acc[16] = {};
  const int ncol0 = wv * 32;

  // ================= phase 1: causal conv as im2col GEMM ==================
  // A[i, tau] = u[b, s0+i-tau] = sUr[127 - i + tau];  B[tau, n] = H[n, tau].
#pragma unroll 2
  for (int k0 = 0; k0 < kmax; k0 += 32) {
    v16bf bt[2];
#pragma unroll
    for (int nt = 0; nt < 2; ++nt) {
      const __bf16* bp = Hb + (size_t)(ncol0 + nt * 16 + lm) * SEQ + k0 + hg * 16;
      bt[nt] = *(const v16bf*)bp;       // 32B aligned: two b128 loads
    }
    // branch-free prefetch of the next Hb k-slab (clamped)
    int k0p = (k0 + 32 < kmax) ? (k0 + 32) : k0;
    __builtin_prefetch(Hb + (size_t)(ncol0 + lm) * SEQ + k0p, 0, 1);

    AB16 at[8];
#pragma unroll
    for (int mt = 0; mt < 8; ++mt) {
      int i = mt * 16 + lm;
#pragma unroll
      for (int v = 0; v < 8; ++v) {
        // A in contiguous k order (same as B): VGPR v holds K pair; lanes 16-31 take K+16.
        int K = 2 * v + 16 * hg;
        int j = (MT - 1) - i + k0 + K;
        unsigned lo = __builtin_bit_cast(unsigned short, sUr[j]);
        unsigned hi = __builtin_bit_cast(unsigned short, sUr[j + 1]);
        at[mt].u[v] = lo | (hi << 16);
      }
    }
#pragma unroll
    for (int mt = 0; mt < 8; ++mt)
#pragma unroll
      for (int nt = 0; nt < 2; ++nt)
        acc[mt * 2 + nt] = __builtin_amdgcn_wmma_f32_16x16x32_bf16(
            false, at[mt].v, false, bt[nt], (short)0, acc[mt * 2 + nt],
            false, false);
  }

  // ======================= phase 2: x @ w_x ===============================
  const float* xb = x + ((size_t)bb * SEQ + s0) * NIN;
  for (int k0 = 0; k0 < NIN; k0 += 64) {
    __syncthreads();
    { // stage 128x64 f32 x-tile as bf16 into LDS
      int r = tid >> 1, seg = (tid & 1) * 32;
      const float4* src = (const float4*)(xb + (size_t)r * NIN + k0 + seg);
      __bf16* dst = sX + r * 72 + seg;
#pragma unroll
      for (int q = 0; q < 8; ++q) {
        float4 f = src[q];
        dst[q * 4 + 0] = f2bf(f.x); dst[q * 4 + 1] = f2bf(f.y);
        dst[q * 4 + 2] = f2bf(f.z); dst[q * 4 + 3] = f2bf(f.w);
      }
    }
    __syncthreads();
#pragma unroll
    for (int kk = 0; kk < 64; kk += 32) {
      v16bf bt[2];
#pragma unroll
      for (int nt = 0; nt < 2; ++nt) {
        const __bf16* bp =
            wxb + (size_t)(ncol0 + nt * 16 + lm) * NIN + k0 + kk + hg * 16;
        bt[nt] = *(const v16bf*)bp;
      }
      AB16 at[8];
#pragma unroll
      for (int mt = 0; mt < 8; ++mt) {
        int i = mt * 16 + lm;
#pragma unroll
        for (int v = 0; v < 8; ++v) {
          int K = 2 * v + 16 * hg;
          at[mt].u[v] = *(const unsigned*)(sX + i * 72 + kk + K); // 4B aligned
        }
      }
#pragma unroll
      for (int mt = 0; mt < 8; ++mt)
#pragma unroll
        for (int nt = 0; nt < 2; ++nt)
          acc[mt * 2 + nt] = __builtin_amdgcn_wmma_f32_16x16x32_bf16(
              false, at[mt].v, false, bt[nt], (short)0, acc[mt * 2 + nt],
              false, false);
    }
  }

  // ================= epilogue: + b_x, relu, store =========================
  // C layout: VGPR r -> M = r (lanes 0-15) / r+8 (lanes 16-31), N = lane&15.
  float* ob = out + ((size_t)bb * SEQ + s0) * NH;
#pragma unroll
  for (int mt = 0; mt < 8; ++mt) {
#pragma unroll
    for (int nt = 0; nt < 2; ++nt) {
      int col = ncol0 + nt * 16 + lm;
      float bx = b_x[col];
      v8f a = acc[mt * 2 + nt];
#pragma unroll
      for (int r = 0; r < 8; ++r) {
        int row = mt * 16 + r + hg * 8;
        ob[(size_t)row * NH + col] = fmaxf(a[r] + bx, 0.f);
      }
    }
  }
}

// ---------------------------------------------------------------------------
extern "C" void kernel_launch(void* const* d_in, const int* in_sizes, int n_in,
                              void* d_out, int out_size, void* d_ws,
                              size_t ws_size, hipStream_t stream) {
  const float* x   = (const float*)d_in[0];
  const float* w_u = (const float*)d_in[1];
  const float* b_u = (const float*)d_in[2];
  const float* w_x = (const float*)d_in[3];
  const float* b_x = (const float*)d_in[4];
  const float* H   = (const float*)d_in[5];
  float* out = (float*)d_out;

  char* ws = (char*)d_ws;
  __bf16* ubf = (__bf16*)ws;                         // 65536 bf16 = 128 KB
  __bf16* Hb  = (__bf16*)(ws + 131072);              // 1Mi  bf16  =   2 MB
  __bf16* wxb = (__bf16*)(ws + 131072 + 2097152);    // 256Ki bf16 = 512 KB

  lmu_u_kernel<<<dim3(BATCHN * SEQ / 8), dim3(256), 0, stream>>>(x, w_u, b_u, ubf);
  lmu_cvt_kernel<<<dim3(NH * SEQ / 256), dim3(256), 0, stream>>>(H, w_x, Hb, wxb);
  lmu_fused_kernel<<<dim3(SEQ / MT, BATCHN), dim3(256), 0, stream>>>(
      x, ubf, Hb, wxb, b_x, out);
}
